// RelationGNN_84997402788219
// MI455X (gfx1250) — matrix operand order repaired and checked by reading, not compiled, not run on hardware
//
#include <hip/hip_runtime.h>
#include <hip/hip_bf16.h>

typedef __attribute__((ext_vector_type(16))) _Float16 v16h;
typedef __attribute__((ext_vector_type(8)))  float    v8f;

constexpr int kNodes   = 50000;
constexpr int kEdges   = 600000;
constexpr int kFeat    = 256;
constexpr int kHid     = 128;
constexpr int kLayers  = 3;
constexpr int kClasses = 32;

// ------------------------- utility kernels -------------------------

__global__ void zero_f32_kernel(float* p, int n4) {
  int i = blockIdx.x * blockDim.x + threadIdx.x;
  int stride = gridDim.x * blockDim.x;
  float4 z = make_float4(0.f, 0.f, 0.f, 0.f);
  for (; i < n4; i += stride) ((float4*)p)[i] = z;
}

// Pack a row-major f32 weight matrix W[K x N] into per-lane WMMA B fragments:
// frag f = kt*(N/16)+nt holds a 32x16 f16 tile; output element i = (f*32+lane)*16 + j,
// with lane = 16*half + n and elem j <-> K = kt*32 + 16*(j>=8) + 8*half + (j%8).
__global__ void pack_b_kernel(const float* __restrict__ W, _Float16* __restrict__ out,
                              int K, int N) {
  int i = blockIdx.x * blockDim.x + threadIdx.x;
  int total = K * N;
  if (i >= total) return;
  int ntiles = N >> 4;
  int f    = i >> 9;          // 512 f16 per fragment (32 lanes x 16)
  int rem  = i & 511;
  int lane = rem >> 4;
  int j    = rem & 15;
  int kt   = f / ntiles;
  int nt   = f - kt * ntiles;
  int n    = lane & 15;
  int half = lane >> 4;
  int Kidx = kt * 32 + (j >> 3) * 16 + half * 8 + (j & 7);
  int col  = nt * 16 + n;
  out[i] = (_Float16)W[Kidx * N + col];
}

// agg[dst[e]] += x[src[e]] : one wave32 per edge, each lane owns 4 floats of the
// 128-float message. Rows are L2-resident (25.6MB << 192MB L2), atomics resolve in L2.
__global__ void scatter_add_kernel(const float* __restrict__ x, const int* __restrict__ src,
                                   const int* __restrict__ dst, float* agg, int nE) {
  int wid  = (blockIdx.x * blockDim.x + threadIdx.x) >> 5;
  int lane = threadIdx.x & 31;
  if (wid >= nE) return;
  int s = src[wid];
  int t = dst[wid];
  float4 v = ((const float4*)(x + (long long)s * kHid))[lane];
  float* o = agg + (long long)t * kHid + lane * 4;
  atomicAdd(o + 0, v.x);
  atomicAdd(o + 1, v.y);
  atomicAdd(o + 2, v.z);
  atomicAdd(o + 3, v.w);
}

// ---------------- CDNA5 async memory->LDS staging ----------------
// Copies `bytes` (multiple of 4096, 16B-aligned) from global `src` into LDS at
// byte offset `lds_base`, using GLOBAL_LOAD_ASYNC_TO_LDS_B128 (ASYNCcnt-tracked,
// no VGPR round trip). All 256 threads of the block participate uniformly.
__device__ __forceinline__ void async_stage(const _Float16* __restrict__ src,
                                            unsigned lds_base, int bytes, int tid) {
  unsigned long long ga = (unsigned long long)(const void*)src;
  for (int off = tid * 16; off < bytes; off += 256 * 16) {
    unsigned l = lds_base + (unsigned)off;
    unsigned long long g = ga + (unsigned long long)(unsigned)off;
    asm volatile("global_load_async_to_lds_b128 %0, %1, off"
                 :: "v"(l), "v"(g)
                 : "memory");
  }
}

__device__ __forceinline__ void async_stage_wait() {
  asm volatile("s_wait_asynccnt 0" ::: "memory");
}

// ---------------- WMMA fragment helpers (CDNA5 wave32 layouts) ----------------
// A 16x32 f16 from f32 source: lane = 16*half + m ; elem j <-> K = 16*(j>=8) + 8*half + (j%8)
__device__ __forceinline__ v16h load_a_frag(const float* A, int ld, int row, int kb, int half) {
  const float* p = A + (long long)row * ld + kb + half * 8;
  float4 x0 = ((const float4*)p)[0];
  float4 x1 = ((const float4*)p)[1];
  float4 y0 = ((const float4*)(p + 16))[0];
  float4 y1 = ((const float4*)(p + 16))[1];
  v16h a;
  a[0]  = (_Float16)x0.x; a[1]  = (_Float16)x0.y; a[2]  = (_Float16)x0.z; a[3]  = (_Float16)x0.w;
  a[4]  = (_Float16)x1.x; a[5]  = (_Float16)x1.y; a[6]  = (_Float16)x1.z; a[7]  = (_Float16)x1.w;
  a[8]  = (_Float16)y0.x; a[9]  = (_Float16)y0.y; a[10] = (_Float16)y0.z; a[11] = (_Float16)y0.w;
  a[12] = (_Float16)y1.x; a[13] = (_Float16)y1.y; a[14] = (_Float16)y1.z; a[15] = (_Float16)y1.w;
  return a;
}

// Packed B fragment from LDS: one 32-byte per-lane read (2 x ds_load_b128).
__device__ __forceinline__ v16h load_b_lds(const _Float16* Fs, int fidx, int lane) {
  return *(const v16h*)(Fs + ((fidx * 32 + lane) << 4));
}

// out[M x 128] = relu(A1[M x K1] @ W1  (+ A2[M x K2] @ W2)  + bias)
// F1/F2 are pre-packed B fragment buffers, staged block-wide into LDS via async
// copy; all 8 waves share them. One wave per 16x128 output strip; `out` may
// alias A1 (each wave reads all of its own A rows before storing).
__global__ void __launch_bounds__(256)
node_gemm_relu_wmma(const float* A1, const _Float16* __restrict__ F1, int K1,
                    const float* A2, const _Float16* __restrict__ F2, int K2,
                    const float* __restrict__ bias, float* out, int M) {
  __shared__ __align__(64) _Float16 smem[32768];   // 64 KB of packed B fragments
  int tid  = threadIdx.x;
  int lane = tid & 31;

  // block-wide async stage of all B fragments for this GEMM (before any exit)
  int bytes1 = (K1 >> 5) * 8 * 1024;               // (K1/32) k-tiles x 8 n-tiles x 1KB
  unsigned lds0 = (unsigned)(size_t)&smem[0];
  async_stage(F1, lds0, bytes1, tid);
  if (A2 != nullptr) {
    int bytes2 = (K2 >> 5) * 8 * 1024;
    async_stage(F2, lds0 + (unsigned)bytes1, bytes2, tid);
  }
  async_stage_wait();
  __syncthreads();

  int wid = (blockIdx.x * blockDim.x + tid) >> 5;
  int strips = M >> 4;
  if (wid >= strips) return;
  int mbase = wid << 4;
  int nl   = lane & 15;
  int half = lane >> 4;
  int arow = mbase + nl;

  const _Float16* F1s = smem;
  const _Float16* F2s = smem + (bytes1 >> 1);

  v8f acc[8];
#pragma unroll
  for (int nt = 0; nt < 8; ++nt) {
    float bv = bias[nt * 16 + nl];
#pragma unroll
    for (int r = 0; r < 8; ++r) acc[nt][r] = bv;
  }

  for (int kb = 0, kt = 0; kb < K1; kb += 32, ++kt) {
    // prefetch next A k-tile of this row (global_prefetch_b8); speculative, drop-safe
    __builtin_prefetch(A1 + (long long)arow * K1 + kb + 32, 0, 1);
    v16h a = load_a_frag(A1, K1, arow, kb, half);
#pragma unroll
    for (int nt = 0; nt < 8; ++nt) {
      v16h b = load_b_lds(F1s, kt * 8 + nt, lane);
      acc[nt] = __builtin_amdgcn_wmma_f32_16x16x32_f16(false, a, false, b,
                                                       (short)0, acc[nt], false, false);
    }
  }
  if (A2 != nullptr) {
    for (int kb = 0, kt = 0; kb < K2; kb += 32, ++kt) {
      v16h a = load_a_frag(A2, K2, arow, kb, half);
#pragma unroll
      for (int nt = 0; nt < 8; ++nt) {
        v16h b = load_b_lds(F2s, kt * 8 + nt, lane);
        acc[nt] = __builtin_amdgcn_wmma_f32_16x16x32_f16(false, a, false, b,
                                                         (short)0, acc[nt], false, false);
      }
    }
  }

#pragma unroll
  for (int nt = 0; nt < 8; ++nt)
#pragma unroll
    for (int r = 0; r < 8; ++r)
      out[(long long)(mbase + half * 8 + r) * kHid + nt * 16 + nl] = fmaxf(acc[nt][r], 0.f);
}

// out[e, :] = concat(x[src[e]], x[dst[e]]) @ head_w + head_b
// Fh = packed head_w[256 x 32] (8 k-tiles x 2 n-tiles = 16 KB), async-staged in LDS.
__global__ void __launch_bounds__(256)
edge_head_wmma(const float* __restrict__ x, const int* __restrict__ src,
               const int* __restrict__ dst, const _Float16* __restrict__ Fh,
               const float* __restrict__ bias, float* __restrict__ out, int nE) {
  __shared__ __align__(64) _Float16 smem[8192];    // 16 KB
  int tid  = threadIdx.x;
  int lane = tid & 31;

  async_stage(Fh, (unsigned)(size_t)&smem[0], 16 * 1024, tid);
  async_stage_wait();
  __syncthreads();

  int wid = (blockIdx.x * blockDim.x + tid) >> 5;
  int tiles = nE >> 4;
  if (wid >= tiles) return;
  int ebase = wid << 4;
  int nl   = lane & 15;
  int half = lane >> 4;
  int erow = ebase + nl;
  int rs = src[erow];
  int rd = dst[erow];

  v8f acc[2];
#pragma unroll
  for (int nt = 0; nt < 2; ++nt) {
    float bv = bias[nt * 16 + nl];
#pragma unroll
    for (int r = 0; r < 8; ++r) acc[nt][r] = bv;
  }

#pragma unroll
  for (int kt = 0; kt < 4; ++kt) {
    v16h a = load_a_frag(x, kHid, rs, kt * 32, half);
#pragma unroll
    for (int nt = 0; nt < 2; ++nt) {
      v16h b = load_b_lds(smem, kt * 2 + nt, lane);
      acc[nt] = __builtin_amdgcn_wmma_f32_16x16x32_f16(false, a, false, b,
                                                       (short)0, acc[nt], false, false);
    }
  }
#pragma unroll
  for (int kt = 0; kt < 4; ++kt) {
    v16h a = load_a_frag(x, kHid, rd, kt * 32, half);
#pragma unroll
    for (int nt = 0; nt < 2; ++nt) {
      v16h b = load_b_lds(smem, (4 + kt) * 2 + nt, lane);
      acc[nt] = __builtin_amdgcn_wmma_f32_16x16x32_f16(false, a, false, b,
                                                       (short)0, acc[nt], false, false);
    }
  }

#pragma unroll
  for (int nt = 0; nt < 2; ++nt)
#pragma unroll
    for (int r = 0; r < 8; ++r)
      out[(long long)(ebase + half * 8 + r) * kClasses + nt * 16 + nl] = acc[nt][r];
}

// ------------------------------ launcher ------------------------------

extern "C" void kernel_launch(void* const* d_in, const int* in_sizes, int n_in,
                              void* d_out, int out_size, void* d_ws, size_t ws_size,
                              hipStream_t stream) {
  (void)in_sizes; (void)n_in; (void)out_size; (void)ws_size;

  const float* obj    = (const float*)d_in[0];
  const int*   eidx   = (const int*)d_in[1];
  const float* emb_w  = (const float*)d_in[2];
  const float* emb_b  = (const float*)d_in[3];
  const float* rel_w  = (const float*)d_in[4];
  const float* rel_b  = (const float*)d_in[5];
  const float* root_w = (const float*)d_in[6];
  const float* head_w = (const float*)d_in[7];
  const float* head_b = (const float*)d_in[8];
  float* out = (float*)d_out;

  // workspace: two ping-pong node buffers + packed f16 weight fragments (~52 MB)
  char* ws = (char*)d_ws;
  size_t nodeBytes = (size_t)kNodes * kHid * sizeof(float);
  float* B0 = (float*)(ws);
  float* B1 = (float*)(ws + nodeBytes);
  _Float16* emb_f  = (_Float16*)(ws + 2 * nodeBytes);       // 256x128 packed
  _Float16* rel_f  = emb_f  + kFeat * kHid;                 // 3 x 128x128 packed
  _Float16* root_f = rel_f  + kLayers * kHid * kHid;
  _Float16* head_f = root_f + kLayers * kHid * kHid;        // 256x32 packed

  // pack weights into WMMA B-fragment layout (tiny one-shot kernels)
  pack_b_kernel<<<(kFeat * kHid + 255) / 256, 256, 0, stream>>>(emb_w, emb_f, kFeat, kHid);
  for (int l = 0; l < kLayers; ++l) {
    pack_b_kernel<<<(kHid * kHid + 255) / 256, 256, 0, stream>>>(
        rel_w + (size_t)l * kHid * kHid, rel_f + (size_t)l * kHid * kHid, kHid, kHid);
    pack_b_kernel<<<(kHid * kHid + 255) / 256, 256, 0, stream>>>(
        root_w + (size_t)l * kHid * kHid, root_f + (size_t)l * kHid * kHid, kHid, kHid);
  }
  pack_b_kernel<<<(kFeat * kClasses + 255) / 256, 256, 0, stream>>>(head_w, head_f, kFeat, kClasses);

  const int strips = kNodes / 16;            // 3125 waves, 8 waves per 256-thread block
  const int gemmBlocks = (strips + 7) / 8;

  // x = relu(obj @ emb_w + emb_b) -> B0
  node_gemm_relu_wmma<<<gemmBlocks, 256, 0, stream>>>(obj, emb_f, kFeat,
                                                      nullptr, nullptr, 0,
                                                      emb_b, B0, kNodes);

  const int* srcI = eidx;
  const int* dstI = eidx + kEdges;

  float* xb = B0;
  float* ab = B1;
  for (int l = 0; l < kLayers; ++l) {
    zero_f32_kernel<<<512, 256, 0, stream>>>(ab, kNodes * kHid / 4);
    scatter_add_kernel<<<kEdges / 8, 256, 0, stream>>>(xb, srcI, dstI, ab, kEdges);
    // x' = relu(agg @ rel_w[l] + rel_b[l] + x @ root_w[l]), written in place over agg
    node_gemm_relu_wmma<<<gemmBlocks, 256, 0, stream>>>(
        ab, rel_f + (size_t)l * kHid * kHid, kHid,
        xb, root_f + (size_t)l * kHid * kHid, kHid,
        rel_b + l * kHid, ab, kNodes);
    float* t = xb; xb = ab; ab = t;
  }

  // edge head: 37500 tiles of 16 edges
  const int etiles = kEdges / 16;
  edge_head_wmma<<<(etiles + 7) / 8, 256, 0, stream>>>(xb, srcI, dstI, head_f, head_b, out, kEdges);
}
